// Hierarchical_Model_89524298318366
// MI455X (gfx1250) — compile-verified, matrix operand order checked by reading, not run on hardware
//
#include <hip/hip_runtime.h>
#include <hip/hip_fp16.h>

typedef __attribute__((ext_vector_type(16))) _Float16 v16h;
typedef __attribute__((ext_vector_type(8)))  float    v8f;
typedef int v4i __attribute__((vector_size(16)));

#define Lg   4096
#define Eg   300
#define EP   320      // E padded to multiple of 32 for WMMA K
#define KT   10       // EP/32 k-tiles
#define Hg   512
#define G4   2048     // 4*H
#define Tg   11
#define START_TAG 9
#define STOP_TAG  10
#define NEG_INF   (-10000.0f)

#define AS1 __attribute__((address_space(1)))
#define AS3 __attribute__((address_space(3)))

// --------------------------- CDNA5 async global->LDS copy (ASYNCcnt path)
#if defined(__has_builtin) && __has_builtin(__builtin_amdgcn_global_load_async_to_lds_b128)
__device__ __forceinline__ void async_copy16(const void* g, void* l) {
    __builtin_amdgcn_global_load_async_to_lds_b128(
        (AS1 v4i*)g, (AS3 v4i*)l, 0, 0);
}
#else
__device__ __forceinline__ void async_copy16(const void* g, void* l) {
    unsigned lds_off = (unsigned)(size_t)(AS3 char*)l;
    asm volatile("global_load_async_to_lds_b128 %0, %1, off"
                 :: "v"(lds_off), "v"((unsigned long long)(size_t)g)
                 : "memory");
}
#endif

#if defined(__has_builtin) && __has_builtin(__builtin_amdgcn_s_wait_asynccnt)
__device__ __forceinline__ void async_wait0() { __builtin_amdgcn_s_wait_asynccnt(0); }
#else
__device__ __forceinline__ void async_wait0() {
    asm volatile("s_wait_asynccnt 0x0" ::: "memory");
}
#endif

// ---------------------------------------------------------------- init sync
__global__ void init_sync_kernel(unsigned* s) {
    if (threadIdx.x < 2) s[threadIdx.x] = 0u;
}

// ------------------------------------------------- fp32 -> fp16 weight pack
// wih layout: [2][G4][EP], zero-padded K in [300,320)
__global__ void conv_wih_kernel(const float* __restrict__ wf,
                                const float* __restrict__ wb,
                                _Float16* __restrict__ wih) {
    int idx = blockIdx.x * 256 + threadIdx.x;
    if (idx >= 2 * G4 * EP) return;
    int dir = idx / (G4 * EP);
    int r   = (idx / EP) % G4;
    int k   = idx % EP;
    const float* w = dir ? wb : wf;
    float v = (k < Eg) ? w[(size_t)r * Eg + k] : 0.0f;
    wih[idx] = (_Float16)v;
}

// -------------------------------------------- embedding gather + fp16 pack
// xh layout: [L][EP], zero-padded K
__global__ void gather_embed_kernel(const int* __restrict__ sent,
                                    const float* __restrict__ emb,
                                    _Float16* __restrict__ xh) {
    int idx = blockIdx.x * 256 + threadIdx.x;
    if (idx >= Lg * EP) return;
    int t = idx / EP;
    int k = idx % EP;
    float v = 0.0f;
    if (k < Eg) v = emb[(size_t)sent[t] * Eg + k];
    xh[idx] = (_Float16)v;
}

// -------------------------------------------------- input projection GEMM
// xg[dir][t][r] = sum_k xh[t][k]*wih[dir][r][k] + b_ih[r] + b_hh[r]
// Block = 8 waves sharing one 16-row A tile (time rows), staged once into
// LDS in *fragment order* via async global->LDS b128 copies; each wave then
// reads its A fragment as two conflict-free ds_load_b128 per k-tile and
// streams its private B tile (gate rows) from global/L2.
__global__ void input_gemm_wmma_kernel(const _Float16* __restrict__ xh,   // [L][EP]
                                       const _Float16* __restrict__ wih,  // [2][G4][EP]
                                       const float* __restrict__ b_ih_f,
                                       const float* __restrict__ b_hh_f,
                                       const float* __restrict__ b_ih_b,
                                       const float* __restrict__ b_hh_b,
                                       float* __restrict__ xg) {          // [2][L][G4]
    // atile[kt][j][lane] : 16-byte chunk = halfs k = kt*32 + j*16 + khi*8 .. +7
    // of time-row m = lane&15  (khi = lane>>4)
    __shared__ _Float16 atile[KT * 512] __attribute__((aligned(16)));

    const int lane  = threadIdx.x & 31;
    const int wave  = threadIdx.x >> 5;
    const int ntile = blockIdx.x * 8 + wave;   // 0..127
    const int mtile = blockIdx.y;              // 0..255
    const int dir   = blockIdx.z;              // 0..1
    const int laneL = lane & 15;
    const int khi   = (lane >> 4) & 1;

    const int m = mtile * 16 + laneL;          // time row (A)
    const int n = ntile * 16 + laneL;          // gate row (B)

    // ---- async-stage the shared A tile (20 chunk groups over 8 waves)
    for (int g = wave; g < 2 * KT; g += 8) {
        int kt = g >> 1, j = g & 1;
        const _Float16* src = xh + (size_t)m * EP + kt * 32 + j * 16 + khi * 8;
        _Float16*       dst = atile + kt * 512 + j * 256 + lane * 8;
        async_copy16(src, dst);
    }
    async_wait0();
    __syncthreads();

    const unsigned* brow = (const unsigned*)(wih + ((size_t)dir * G4 + n) * EP);

    v8f acc = {};
    for (int kt = 0; kt < KT; ++kt) {
        union { v16h v; uint4 q[2]; } A;
        union { v16h v; unsigned u[8]; } B;
        A.q[0] = *(const uint4*)(atile + kt * 512 + lane * 8);
        A.q[1] = *(const uint4*)(atile + kt * 512 + 256 + lane * 8);
#pragma unroll
        for (int v = 0; v < 8; ++v) {
            int kb = ((v & 3) << 1) + ((v >> 2) << 4) + (khi << 3);  // even
            B.u[v] = brow[(kt * 32 + kb) >> 1];
        }
        acc = __builtin_amdgcn_wmma_f32_16x16x32_f16(
            /*neg_a=*/false, A.v, /*neg_b=*/false, B.v,
            /*c_mod=*/(short)0, acc, /*reuse_a=*/false, /*reuse_b=*/false);
    }

    const float* bi = dir ? b_ih_b : b_ih_f;
    const float* bh = dir ? b_hh_b : b_hh_f;
    float bias = bi[n] + bh[n];
    float* outd = xg + (size_t)dir * Lg * G4;
#pragma unroll
    for (int v = 0; v < 8; ++v) {
        int tt = mtile * 16 + v + khi * 8;     // C layout: VGPR v -> M=v (+8 upper half)
        outd[(size_t)tt * G4 + n] = acc[v] + bias;
    }
}

// ----------------------------------------------------- cross-WGP barrier
__device__ __forceinline__ void grid_barrier(unsigned* cnt, unsigned* gen,
                                             unsigned nwg) {
    __syncthreads();
    __builtin_amdgcn_s_cluster_barrier();   // s_barrier_signal/-wait -3 (NOP if unclustered)
    if (threadIdx.x == 0) {
        unsigned g = __hip_atomic_load(gen, __ATOMIC_ACQUIRE, __HIP_MEMORY_SCOPE_AGENT);
        unsigned a = __hip_atomic_fetch_add(cnt, 1u, __ATOMIC_ACQ_REL, __HIP_MEMORY_SCOPE_AGENT);
        if (a == nwg - 1u) {
            __hip_atomic_store(cnt, 0u, __ATOMIC_RELAXED, __HIP_MEMORY_SCOPE_AGENT);
            __hip_atomic_store(gen, g + 1u, __ATOMIC_RELEASE, __HIP_MEMORY_SCOPE_AGENT);
        } else {
            while (__hip_atomic_load(gen, __ATOMIC_ACQUIRE, __HIP_MEMORY_SCOPE_AGENT) == g)
                __builtin_amdgcn_s_sleep(1);
        }
    }
    __syncthreads();
}

// ------------------------------------------------------- LSTM recurrence
// 32 workgroups: dir = blk>>4, part = blk&15. Each WG owns 32 hidden units
// (128 gate rows). Weights live in LDS (256KB, float4-interleaved so the hot
// loop is one ds_load_b128 per lane per 4 k's), h exchanged via L2 with a
// parity-double-buffered hbuf and one cross-WGP barrier per step.
__global__ void lstm_recur_kernel(const float* __restrict__ xg,     // [2][L][G4]
                                  const float* __restrict__ whh_f,  // [G4][H]
                                  const float* __restrict__ whh_b,  // [G4][H]
                                  float* hbuf,                      // [2][2][H]
                                  float* h_all,                     // [2][L][H]
                                  unsigned* sync) {
    const int tid  = threadIdx.x;        // 0..127
    const int dir  = blockIdx.x >> 4;
    const int part = blockIdx.x & 15;
    const float* whh = dir ? whh_b : whh_f;
    const float* xgd = xg + (size_t)dir * Lg * G4;
    float* hb = hbuf + (size_t)dir * 2 * Hg;
    float* ha = h_all + (size_t)dir * Lg * Hg;

    extern __shared__ float smem[];
    float4* w_s4 = (float4*)smem;            // [H/4][128]  w_s4[k4*128+tid] = whh[row][4k4..4k4+3]
    float*  h_s  = smem + Hg * 128;          // [H]
    float*  g_s  = h_s + Hg;                 // [128]

    const int gate = tid >> 5;                 // 0..3 (i,f,g,o)
    const int ul   = tid & 31;                 // local unit
    const int row  = gate * Hg + part * 32 + ul;

    // stage this thread's gate-row into LDS, float4-interleaved
    const float4* wr = (const float4*)(whh + (size_t)row * Hg);
    for (int k4 = 0; k4 < Hg / 4; ++k4)
        w_s4[k4 * 128 + tid] = wr[k4];

    if (part == 0) {   // zero both parity buffers of h
        for (int i = tid; i < 2 * Hg; i += 128)
            __hip_atomic_store(&hb[i], 0.0f, __ATOMIC_RELEASE, __HIP_MEMORY_SCOPE_AGENT);
    }
    float c = 0.0f;
    const float4* h_s4 = (const float4*)h_s;

    for (int s = 0; s < Lg; ++s) {
        grid_barrier(sync, sync + 1, 32u);     // h_{s-1} globally visible
        const int t = dir ? (Lg - 1 - s) : s;
        for (int i = tid; i < Hg; i += 128)
            h_s[i] = __hip_atomic_load(&hb[(s & 1) * Hg + i],
                                       __ATOMIC_ACQUIRE, __HIP_MEMORY_SCOPE_AGENT);
        __syncthreads();
        float acc = xgd[(size_t)t * G4 + row];
#pragma unroll 8
        for (int k4 = 0; k4 < Hg / 4; ++k4) {
            float4 w = w_s4[k4 * 128 + tid];
            float4 h = h_s4[k4];
            acc = fmaf(w.x, h.x, acc);
            acc = fmaf(w.y, h.y, acc);
            acc = fmaf(w.z, h.z, acc);
            acc = fmaf(w.w, h.w, acc);
        }
        g_s[tid] = acc;
        __syncthreads();
        if (tid < 32) {
            float gi = g_s[tid];
            float gf = g_s[32 + tid];
            float gg = g_s[64 + tid];
            float go = g_s[96 + tid];
            float ig = 1.0f / (1.0f + expf(-gi));
            float fg = 1.0f / (1.0f + expf(-gf));
            float og = 1.0f / (1.0f + expf(-go));
            c = fg * c + ig * tanhf(gg);
            float h = og * tanhf(c);
            int u = part * 32 + tid;
            __hip_atomic_store(&hb[((s + 1) & 1) * Hg + u], h,
                               __ATOMIC_RELEASE, __HIP_MEMORY_SCOPE_AGENT);
            ha[(size_t)t * Hg + u] = h;
        }
        __syncthreads();
    }
}

// ---------------------------------------------------------- tag projection
__global__ void tag_proj_kernel(const float* __restrict__ h_all,  // [2][L][H]
                                const float* __restrict__ w_tag,  // [T][2H]
                                const float* __restrict__ b_tag,  // [T]
                                float* __restrict__ feats) {      // [L][T]
    int idx = blockIdx.x * 256 + threadIdx.x;
    int t   = idx >> 4;
    int tag = idx & 15;
    if (t >= Lg || tag >= Tg) return;
    const float* hf  = h_all + (size_t)t * Hg;
    const float* hbk = h_all + (size_t)Lg * Hg + (size_t)t * Hg;
    const float* wt  = w_tag + (size_t)tag * (2 * Hg);
    float acc = b_tag[tag];
    const float4* hf4 = (const float4*)hf;
    const float4* hb4 = (const float4*)hbk;
    const float4* w0  = (const float4*)wt;
    const float4* w1  = (const float4*)(wt + Hg);
    for (int k = 0; k < Hg / 4; ++k) {
        float4 a = hf4[k], w = w0[k];
        acc = fmaf(a.x, w.x, acc); acc = fmaf(a.y, w.y, acc);
        acc = fmaf(a.z, w.z, acc); acc = fmaf(a.w, w.w, acc);
        float4 b = hb4[k], v = w1[k];
        acc = fmaf(b.x, v.x, acc); acc = fmaf(b.y, v.y, acc);
        acc = fmaf(b.z, v.z, acc); acc = fmaf(b.w, v.w, acc);
    }
    feats[t * Tg + tag] = acc;
}

// ------------------------------------------------------------------ viterbi
__global__ void viterbi_kernel(const float* __restrict__ feats,   // [L][T]
                               const float* __restrict__ trans,   // [T][T]
                               int* __restrict__ bptr,            // [L][T]
                               float* __restrict__ out) {         // [1 + L]
    __shared__ float fv[Tg];
    __shared__ float fvn[Tg];
    int lane = threadIdx.x;
    if (lane < Tg) fv[lane] = (lane == START_TAG) ? 0.0f : NEG_INF;
    __syncthreads();
    for (int t = 0; t < Lg; ++t) {
        if (lane < Tg) {
            float best = -3.4e38f; int bi = 0;
            for (int tp = 0; tp < Tg; ++tp) {
                float sc = fv[tp] + trans[lane * Tg + tp];
                if (sc > best) { best = sc; bi = tp; }
            }
            bptr[t * Tg + lane] = bi;
            fvn[lane] = best + feats[t * Tg + lane];
        }
        __syncthreads();
        if (lane < Tg) fv[lane] = fvn[lane];
        __syncthreads();
    }
    if (lane == 0) {
        float best = -3.4e38f; int bi = 0;
        for (int tn = 0; tn < Tg; ++tn) {
            float sc = fv[tn] + trans[STOP_TAG * Tg + tn];
            if (sc > best) { best = sc; bi = tn; }
        }
        out[0] = best;
        int cur = bi;
        out[1 + (Lg - 1)] = (float)cur;
        for (int t = Lg - 2; t >= 0; --t) {
            cur = bptr[(t + 1) * Tg + cur];
            out[1 + t] = (float)cur;
        }
    }
}

// ------------------------------------------------------------------ launch
static inline size_t align256(size_t x) { return (x + 255) & ~(size_t)255; }

extern "C" void kernel_launch(void* const* d_in, const int* in_sizes, int n_in,
                              void* d_out, int out_size, void* d_ws, size_t ws_size,
                              hipStream_t stream) {
    const int*   sentence  = (const int*)  d_in[0];
    const float* embedding = (const float*)d_in[1];
    const float* w_ih_f    = (const float*)d_in[2];
    const float* w_hh_f    = (const float*)d_in[3];
    const float* b_ih_f    = (const float*)d_in[4];
    const float* b_hh_f    = (const float*)d_in[5];
    const float* w_ih_b    = (const float*)d_in[6];
    const float* w_hh_b    = (const float*)d_in[7];
    const float* b_ih_b    = (const float*)d_in[8];
    const float* b_hh_b    = (const float*)d_in[9];
    const float* w_tag     = (const float*)d_in[10];
    const float* b_tag     = (const float*)d_in[11];
    const float* trans     = (const float*)d_in[12];
    float* out = (float*)d_out;

    char* ws = (char*)d_ws;
    size_t off = 0;
    _Float16* xh   = (_Float16*)(ws + off); off = align256(off + (size_t)Lg * EP * 2);
    _Float16* wih  = (_Float16*)(ws + off); off = align256(off + (size_t)2 * G4 * EP * 2);
    float*    xg   = (float*)   (ws + off); off = align256(off + (size_t)2 * Lg * G4 * 4);
    float*    hall = (float*)   (ws + off); off = align256(off + (size_t)2 * Lg * Hg * 4);
    float*    hbuf = (float*)   (ws + off); off = align256(off + (size_t)2 * 2 * Hg * 4);
    unsigned* sync = (unsigned*)(ws + off); off = align256(off + 256);
    float*    feats= (float*)   (ws + off); off = align256(off + (size_t)Lg * Tg * 4);
    int*      bptr = (int*)     (ws + off); off = align256(off + (size_t)Lg * Tg * 4);

    init_sync_kernel<<<1, 32, 0, stream>>>(sync);
    conv_wih_kernel<<<(2 * G4 * EP + 255) / 256, 256, 0, stream>>>(w_ih_f, w_ih_b, wih);
    gather_embed_kernel<<<(Lg * EP + 255) / 256, 256, 0, stream>>>(sentence, embedding, xh);

    dim3 ggrid(16, Lg / 16, 2);   // ntile-groups, mtiles, dirs
    input_gemm_wmma_kernel<<<ggrid, 256, 0, stream>>>(xh, wih, b_ih_f, b_hh_f,
                                                      b_ih_b, b_hh_b, xg);

    const int recur_smem = (Hg * 128 + Hg + 128) * (int)sizeof(float);  // ~259KB
    (void)hipFuncSetAttribute((const void*)lstm_recur_kernel,
                              hipFuncAttributeMaxDynamicSharedMemorySize, recur_smem);
    lstm_recur_kernel<<<32, 128, recur_smem, stream>>>(xg, w_hh_f, w_hh_b,
                                                       hbuf, hall, sync);

    tag_proj_kernel<<<(Lg * 16 + 255) / 256, 256, 0, stream>>>(hall, w_tag, b_tag, feats);
    viterbi_kernel<<<1, 32, 0, stream>>>(feats, trans, bptr, out);
}
